// Attention_61117384622245
// MI455X (gfx1250) — matrix operand order repaired and checked by reading, not compiled, not run on hardware
//
#include <hip/hip_runtime.h>
#include <hip/hip_bf16.h>

typedef __bf16 bf16;
typedef __attribute__((ext_vector_type(16))) __bf16 v16bf;
typedef __attribute__((ext_vector_type(8)))  __bf16 v8bf;
typedef __attribute__((ext_vector_type(8)))  float  v8f;

#define CDIM   128
#define L_DIM  8192
#define BATCH  4
#define MROWS  (BATCH * L_DIM)   // 32768
#define NNBR   16
#define NHEAD  8
#define HD     16

// ---------------------------------------------------------------- weights f32 -> bf16
__global__ __launch_bounds__(256) void wcvt_kernel(
    const float* __restrict__ w0, const float* __restrict__ w1,
    const float* __restrict__ w2, const float* __restrict__ w3,
    const float* __restrict__ w4, const float* __restrict__ w5,
    bf16* __restrict__ dst) {
    int gid = blockIdx.x * blockDim.x + threadIdx.x;
    int wi = gid >> 14;        // / 16384
    int e  = gid & 16383;
    const float* src;
    if      (wi == 0) src = w0;
    else if (wi == 1) src = w1;
    else if (wi == 2) src = w2;
    else if (wi == 3) src = w3;
    else if (wi == 4) src = w4;
    else              src = w5;
    dst[gid] = (bf16)src[e];
}

// ---------------------------------------------------------------- LN1: xn = LN(x0)*g1+b1 -> bf16
__global__ __launch_bounds__(128) void ln_kernel(
    const float* __restrict__ x0, const float* __restrict__ g,
    const float* __restrict__ b, bf16* __restrict__ xn) {
    int row = blockIdx.x;
    int t   = threadIdx.x;               // 0..127
    __shared__ float red[CDIM];
    __shared__ float smu, srs;
    float xv = x0[(size_t)row * CDIM + t];
    red[t] = xv; __syncthreads();
    #pragma unroll
    for (int s = 64; s > 0; s >>= 1) { if (t < s) red[t] += red[t + s]; __syncthreads(); }
    if (t == 0) smu = red[0] * (1.0f / CDIM);
    __syncthreads();
    float d = xv - smu;
    red[t] = d * d; __syncthreads();
    #pragma unroll
    for (int s = 64; s > 0; s >>= 1) { if (t < s) red[t] += red[t + s]; __syncthreads(); }
    if (t == 0) srs = rsqrtf(red[0] * (1.0f / CDIM) + 1e-5f);
    __syncthreads();
    xn[(size_t)row * CDIM + t] = (bf16)(d * srs * g[t] + b[t]);
}

// ---------------------------------------------------------------- WMMA tile loads
// A tile 16x32 (MxK), row-major src, ld = CDIM.
// lanes 0-15: m=lane,  K = {k0..k0+15}
// lanes16-31: m=lane-16, K = {k0+16..k0+31}
__device__ __forceinline__ v16bf load_A(const bf16* __restrict__ X, int row0, int k0) {
    int lane = threadIdx.x;
    int m = lane & 15, half = lane >> 4;
    const bf16* p = X + (size_t)(row0 + m) * CDIM + k0 + half * 16;
    v8bf lo = *(const v8bf*)(p);
    v8bf hi = *(const v8bf*)(p + 8);
    v16bf a;
    #pragma unroll
    for (int i = 0; i < 8; ++i) { a[i] = lo[i]; a[i + 8] = hi[i]; }
    return a;
}

// B tile 32x16 (KxN) for Y = X @ W^T  (B[k,n] = W[n,k], W row-major [C,C]).
// lane holds column n = lane&15, K = k0 + (lane>>4)*16 + {0..15}: contiguous 32 B of W row.
__device__ __forceinline__ v16bf load_B(const bf16* __restrict__ W, int n0, int k0) {
    int lane = threadIdx.x;
    int n = lane & 15, g = lane >> 4;
    return *(const v16bf*)(W + (size_t)(n0 + n) * CDIM + k0 + g * 16);
}

// ---------------------------------------------------------------- fused QKV GEMM (shared A tile)
// q -> f32 (precision for scores), k/v -> bf16 (halves the gather traffic)
__global__ __launch_bounds__(32) void gemm_qkv_kernel(
    const bf16* __restrict__ X,
    const bf16* __restrict__ Wq, const bf16* __restrict__ Wk, const bf16* __restrict__ Wv,
    float* __restrict__ qf, bf16* __restrict__ kbf, bf16* __restrict__ vbf) {
    int row0 = blockIdx.x * 16;
    int lane = threadIdx.x;
    v16bf a[4];
    #pragma unroll
    for (int kk = 0; kk < 4; ++kk) a[kk] = load_A(X, row0, kk * 32);

    int col_in = lane & 15;
    int rbase  = (lane >> 4) * 8;
    #pragma unroll
    for (int w = 0; w < 3; ++w) {
        const bf16* W = (w == 0) ? Wq : (w == 1) ? Wk : Wv;
        #pragma unroll
        for (int nt = 0; nt < 8; ++nt) {
            v8f acc = {};
            #pragma unroll
            for (int kk = 0; kk < 4; ++kk) {
                v16bf b = load_B(W, nt * 16, kk * 32);
                acc = __builtin_amdgcn_wmma_f32_16x16x32_bf16(
                    false, a[kk], false, b, (short)0, acc, false, false);
            }
            size_t base = (size_t)(row0 + rbase) * CDIM + nt * 16 + col_in;
            if (w == 0) {
                float* p = qf + base;
                #pragma unroll
                for (int i = 0; i < 8; ++i) p[i * CDIM] = acc[i];
            } else {
                bf16* p = ((w == 1) ? kbf : vbf) + base;
                #pragma unroll
                for (int i = 0; i < 8; ++i) p[i * CDIM] = (bf16)acc[i];
            }
        }
    }
}

// ---------------------------------------------------------------- generic 16-row GEMM, C=K=N=128
// MODE 1: out_bf = bf16(addsrc + X@W^T)    (residual)
// MODE 2: out_bf = bf16(relu(X@W^T))
template <int MODE>
__global__ __launch_bounds__(32) void gemm_kernel(
    const bf16* __restrict__ X, const bf16* __restrict__ W,
    const float* __restrict__ addsrc, bf16* __restrict__ out_bf) {
    int row0 = blockIdx.x * 16;
    int lane = threadIdx.x;
    v16bf a[4];
    #pragma unroll
    for (int kk = 0; kk < 4; ++kk) a[kk] = load_A(X, row0, kk * 32);

    int col_in = lane & 15;
    int rbase  = (lane >> 4) * 8;
    #pragma unroll
    for (int nt = 0; nt < 8; ++nt) {
        v8f acc = {};
        #pragma unroll
        for (int kk = 0; kk < 4; ++kk) {
            v16bf b = load_B(W, nt * 16, kk * 32);
            acc = __builtin_amdgcn_wmma_f32_16x16x32_bf16(
                false, a[kk], false, b, (short)0, acc, false, false);
        }
        size_t base = (size_t)(row0 + rbase) * CDIM + nt * 16 + col_in;
        if (MODE == 1) {
            const float* r = addsrc + base;
            bf16* p = out_bf + base;
            #pragma unroll
            for (int i = 0; i < 8; ++i) p[i * CDIM] = (bf16)(r[i * CDIM] + acc[i]);
        } else {
            bf16* p = out_bf + base;
            #pragma unroll
            for (int i = 0; i < 8; ++i) p[i * CDIM] = (bf16)fmaxf(acc[i], 0.0f);
        }
    }
}

// ---------------------------------------------------------------- gathered neighbor attention
// block = one (b,l) row, 128 threads: t=(h,j) for scores/softmax, t=(h,d) for output
// k/v read as bf16 (halves the dominant L2 gather traffic), math in f32
__global__ __launch_bounds__(128) void attn_kernel(
    const float* __restrict__ q, const bf16* __restrict__ k,
    const bf16* __restrict__ v, const int* __restrict__ query,
    bf16* __restrict__ out) {
    int row = blockIdx.x;            // b*L + l
    int b   = row >> 13;             // / 8192
    int t   = threadIdx.x;           // 0..127
    int h   = t >> 4, j = t & 15;
    __shared__ int   sidx[NNBR];
    __shared__ float sc[NHEAD][NNBR];
    __shared__ float se[NHEAD][NNBR];
    if (t < NNBR) sidx[t] = query[(size_t)row * NNBR + t];
    __syncthreads();

    const float* qr = q + (size_t)row * CDIM + h * HD;
    v16bf kv = *(const v16bf*)(k + (size_t)(b * L_DIM + sidx[j]) * CDIM + h * HD); // 32B aligned
    float s = 0.f;
    #pragma unroll
    for (int d = 0; d < HD; ++d) s += qr[d] * (float)kv[d];
    s *= 0.25f;                      // 1/sqrt(16)
    sc[h][j] = s;
    __syncthreads();

    float mx = sc[h][0];
    #pragma unroll
    for (int jj = 1; jj < NNBR; ++jj) mx = fmaxf(mx, sc[h][jj]);
    se[h][j] = __expf(s - mx);
    __syncthreads();

    float ssum = 0.f;
    #pragma unroll
    for (int jj = 0; jj < NNBR; ++jj) ssum += se[h][jj];
    float inv = 1.f / ssum;

    int d = j;                       // output element (h, d); col == t
    float o = 0.f;
    #pragma unroll
    for (int jj = 0; jj < NNBR; ++jj) {
        o += se[h][jj] * (float)v[(size_t)(b * L_DIM + sidx[jj]) * CDIM + h * HD + d];
    }
    out[(size_t)row * CDIM + t] = (bf16)(o * inv);
}

// ---------------------------------------------------------------- W2 GEMM + LayerNorm + residual
__global__ __launch_bounds__(32) void gemm_ln_kernel(
    const bf16* __restrict__ X, const bf16* __restrict__ W,
    const float* __restrict__ x0,
    const float* __restrict__ g2, const float* __restrict__ b2,
    float* __restrict__ y) {
    int row0 = blockIdx.x * 16;
    int lane = threadIdx.x;
    __shared__ float tile[16][CDIM + 1];   // +1 pad: avoid 64-bank conflicts on column reads

    v16bf a[4];
    #pragma unroll
    for (int kk = 0; kk < 4; ++kk) a[kk] = load_A(X, row0, kk * 32);

    int col_in = lane & 15;
    int rbase  = (lane >> 4) * 8;
    #pragma unroll
    for (int nt = 0; nt < 8; ++nt) {
        v8f acc = {};
        #pragma unroll
        for (int kk = 0; kk < 4; ++kk) {
            v16bf b = load_B(W, nt * 16, kk * 32);
            acc = __builtin_amdgcn_wmma_f32_16x16x32_bf16(
                false, a[kk], false, b, (short)0, acc, false, false);
        }
        #pragma unroll
        for (int i = 0; i < 8; ++i)
            tile[rbase + i][nt * 16 + col_in] = acc[i];
    }
    __syncthreads();

    if (lane < 16) {
        int r = lane;
        float mu = 0.f;
        for (int c = 0; c < CDIM; ++c) mu += tile[r][c];
        mu *= (1.0f / CDIM);
        float var = 0.f;
        for (int c = 0; c < CDIM; ++c) { float d = tile[r][c] - mu; var += d * d; }
        float rs = rsqrtf(var * (1.0f / CDIM) + 1e-5f);
        size_t base = (size_t)(row0 + r) * CDIM;
        for (int c = 0; c < CDIM; ++c)
            y[base + c] = x0[base + c] + (tile[r][c] - mu) * rs * g2[c] + b2[c];
    }
}

// ---------------------------------------------------------------- launch
extern "C" void kernel_launch(void* const* d_in, const int* in_sizes, int n_in,
                              void* d_out, int out_size, void* d_ws, size_t ws_size,
                              hipStream_t stream) {
    const float* x0    = (const float*)d_in[0];
    const int*   query = (const int*)  d_in[1];
    const float* Wq    = (const float*)d_in[2];
    const float* Wk    = (const float*)d_in[3];
    const float* Wv    = (const float*)d_in[4];
    const float* Wm    = (const float*)d_in[5];
    const float* W1    = (const float*)d_in[6];
    const float* W2    = (const float*)d_in[7];
    const float* g1    = (const float*)d_in[8];
    const float* b1    = (const float*)d_in[9];
    const float* g2    = (const float*)d_in[10];
    const float* b2    = (const float*)d_in[11];
    float* out = (float*)d_out;

    char*  ws  = (char*)d_ws;
    size_t off = 0;
    bf16*  wbf  = (bf16*)(ws + off);  off += (size_t)6 * 16384 * 2;
    bf16*  xn   = (bf16*)(ws + off);  off += (size_t)MROWS * CDIM * 2;
    float* qf   = (float*)(ws + off); off += (size_t)MROWS * CDIM * 4;
    bf16*  kbf  = (bf16*)(ws + off);  off += (size_t)MROWS * CDIM * 2;
    bf16*  vbf  = (bf16*)(ws + off);  off += (size_t)MROWS * CDIM * 2;
    bf16*  attn = (bf16*)(ws + off);  off += (size_t)MROWS * CDIM * 2;
    bf16*  msg  = (bf16*)(ws + off);  off += (size_t)MROWS * CDIM * 2;
    bf16*  h1   = (bf16*)(ws + off);  off += (size_t)MROWS * CDIM * 2;

    const int MT = MROWS / 16;   // 2048 tiles

    wcvt_kernel<<<(6 * 16384) / 256, 256, 0, stream>>>(Wq, Wk, Wv, Wm, W1, W2, wbf);
    ln_kernel<<<MROWS, 128, 0, stream>>>(x0, g1, b1, xn);

    gemm_qkv_kernel<<<MT, 32, 0, stream>>>(xn, wbf + 0 * 16384, wbf + 1 * 16384,
                                           wbf + 2 * 16384, qf, kbf, vbf);

    attn_kernel<<<MROWS, 128, 0, stream>>>(qf, kbf, vbf, query, attn);

    gemm_kernel<1><<<MT, 32, 0, stream>>>(attn, wbf + 3 * 16384, x0, msg);
    gemm_kernel<2><<<MT, 32, 0, stream>>>(msg,  wbf + 4 * 16384, nullptr, h1);
    gemm_ln_kernel<<<MT, 32, 0, stream>>>(h1, wbf + 5 * 16384, x0, g2, b2, out);
}
